// TrajectoryGenerator_85375359910307
// MI455X (gfx1250) — matrix operand order repaired and checked by reading, not compiled, not run on hardware
//
#include <hip/hip_runtime.h>
#include <hip/hip_bf16.h>
#include <math.h>

typedef __attribute__((ext_vector_type(16))) __bf16 v16bf;
typedef __attribute__((ext_vector_type(8)))  float  v8f;

#define N_AG 512
#define HID  16

// ---- workspace layout (float offsets) ----
#define WS_H     0        // [512,16] pred_h / h_enc
#define WS_C     8192     // [512,16] pred_c
#define WS_CTX   16384    // [512,16] ctx (pn pooling result, decoder carry)
#define WS_RG    24576    // [512,16] rel_goal
#define WS_POS   32768    // [512,2]
#define WS_POSS  33792    // [512,2] pos_s
#define WS_OUT   34816    // [512,2] output carry (out_pred)
#define WS_SAMP  35840    // [512,2] sampled (= mu)
#define WS_CTXC  36864    // [512,16] ctx_c
#define WS_WCPN  45056    // [2,128] folded corr weights (pn)
#define WS_CBPN  45312    // [128]   folded bias (pn)
#define WS_WCPNC 45440    // [2,128] (pnc)
#define WS_CBPNC 45696    // [128]
#define WS_HJ    45824    // [512,128] per-j hidden contribution (reused pn/pnc)

__device__ __forceinline__ float sigm(float x) { return 1.0f / (1.0f + expf(-x)); }

// ---------------- init: rel_goal, pos, output, ctx=0, c=0 ----------------
__global__ __launch_bounds__(256) void tg_init(
    const float* __restrict__ traj_rel, const float* __restrict__ obs_pos,
    const float* __restrict__ goal, const float* __restrict__ goalW,
    const float* __restrict__ goalB, float* __restrict__ ws)
{
    int i = blockIdx.x * 256 + threadIdx.x;
    if (i >= N_AG) return;
    float lx = obs_pos[7 * N_AG * 2 + i * 2 + 0];
    float ly = obs_pos[7 * N_AG * 2 + i * 2 + 1];
    float rx = goal[i * 2 + 0] - lx;
    float ry = goal[i * 2 + 1] - ly;
    #pragma unroll
    for (int m = 0; m < HID; ++m)
        ws[WS_RG + i * HID + m] = goalW[m * 2] * rx + goalW[m * 2 + 1] * ry + goalB[m];
    ws[WS_POS + i * 2 + 0] = lx;
    ws[WS_POS + i * 2 + 1] = ly;
    ws[WS_OUT + i * 2 + 0] = traj_rel[7 * N_AG * 2 + i * 2 + 0];
    ws[WS_OUT + i * 2 + 1] = traj_rel[7 * N_AG * 2 + i * 2 + 1];
    #pragma unroll
    for (int k = 0; k < HID; ++k) {
        ws[WS_CTX + i * HID + k] = 0.f;
        ws[WS_C   + i * HID + k] = 0.f;
    }
}

// ---------------- fold: Wc = spW^T @ W1a^T, cb = b1 + spb @ W1a^T --------
__global__ __launch_bounds__(256) void tg_fold(
    const float* __restrict__ spW_pn,  const float* __restrict__ spb_pn,
    const float* __restrict__ W1_pn,   const float* __restrict__ b1_pn,
    const float* __restrict__ spW_pnc, const float* __restrict__ spb_pnc,
    const float* __restrict__ W1_pnc,  const float* __restrict__ b1_pnc,
    float* __restrict__ ws)
{
    int t = threadIdx.x;
    const float *spW, *spb, *W1, *b1;
    float *Wc, *cb;
    int m;
    if (t < 128) { spW = spW_pn;  spb = spb_pn;  W1 = W1_pn;  b1 = b1_pn;
                   Wc = ws + WS_WCPN;  cb = ws + WS_CBPN;  m = t; }
    else         { spW = spW_pnc; spb = spb_pnc; W1 = W1_pnc; b1 = b1_pnc;
                   Wc = ws + WS_WCPNC; cb = ws + WS_CBPNC; m = t - 128; }
    float w0 = 0.f, w1 = 0.f, c = b1[m];
    #pragma unroll
    for (int k = 0; k < 16; ++k) {
        float w = W1[m * 32 + k];
        w0 += spW[k * 2 + 0] * w;
        w1 += spW[k * 2 + 1] * w;
        c  += spb[k] * w;
    }
    Wc[m] = w0; Wc[128 + m] = w1; cb[m] = c;
}

// ---------------- encoder: 8 LSTM steps over all agents -----------------
__global__ __launch_bounds__(256) void tg_enc(
    const float* __restrict__ traj_rel, const float* __restrict__ h0,
    const float* __restrict__ c0, const float* __restrict__ encW,
    const float* __restrict__ encB,
    const float* __restrict__ Wih, const float* __restrict__ bih,
    const float* __restrict__ Whh, const float* __restrict__ bhh,
    float* __restrict__ ws)
{
    __shared__ float sWih[1024], sWhh[1024], sbi[64], sbh[64], sEW[32], sEB[16];
    for (int idx = threadIdx.x; idx < 1024; idx += 256) { sWih[idx] = Wih[idx]; sWhh[idx] = Whh[idx]; }
    if (threadIdx.x < 64) { sbi[threadIdx.x] = bih[threadIdx.x]; sbh[threadIdx.x] = bhh[threadIdx.x]; }
    if (threadIdx.x < 32) sEW[threadIdx.x] = encW[threadIdx.x];
    if (threadIdx.x < 16) sEB[threadIdx.x] = encB[threadIdx.x];
    __syncthreads();
    int i = blockIdx.x * 256 + threadIdx.x;
    float h[16], c[16];
    #pragma unroll
    for (int k = 0; k < 16; ++k) { h[k] = h0[i * 16 + k]; c[k] = c0[i * 16 + k]; }
    for (int t = 0; t < 8; ++t) {
        float x0 = traj_rel[(t * N_AG + i) * 2 + 0];
        float x1 = traj_rel[(t * N_AG + i) * 2 + 1];
        float e[16];
        #pragma unroll
        for (int k = 0; k < 16; ++k)
            e[k] = fmaxf(sEW[k * 2] * x0 + sEW[k * 2 + 1] * x1 + sEB[k], 0.f);
        float nh[16], nc[16];
        for (int u = 0; u < 16; ++u) {
            float gi = sbi[u]      + sbh[u];
            float gf = sbi[16 + u] + sbh[16 + u];
            float gg = sbi[32 + u] + sbh[32 + u];
            float go = sbi[48 + u] + sbh[48 + u];
            #pragma unroll
            for (int k = 0; k < 16; ++k) {
                gi += sWih[u * 16 + k]        * e[k] + sWhh[u * 16 + k]        * h[k];
                gf += sWih[(16 + u) * 16 + k] * e[k] + sWhh[(16 + u) * 16 + k] * h[k];
                gg += sWih[(32 + u) * 16 + k] * e[k] + sWhh[(32 + u) * 16 + k] * h[k];
                go += sWih[(48 + u) * 16 + k] * e[k] + sWhh[(48 + u) * 16 + k] * h[k];
            }
            nc[u] = sigm(gf) * c[u] + sigm(gi) * tanhf(gg);
            nh[u] = sigm(go) * tanhf(nc[u]);
        }
        for (int k = 0; k < 16; ++k) { h[k] = nh[k]; c[k] = nc[k]; }
    }
    #pragma unroll
    for (int k = 0; k < 16; ++k) ws[WS_H + i * 16 + k] = h[k];
}

// ---------------- decoder cell: e -> LSTM -> pred_h/c, Hj(pn) -----------
__global__ __launch_bounds__(256) void tg_dec(
    const float* __restrict__ decW, const float* __restrict__ decB,
    const float* __restrict__ Wih, const float* __restrict__ bih,
    const float* __restrict__ Whh, const float* __restrict__ bhh,
    const float* __restrict__ W1, const float* __restrict__ cb,
    float* __restrict__ ws)
{
    __shared__ float sWih[1024], sWhh[1024], sbi[64], sbh[64];
    __shared__ float sDW[288], sDB[16], sW1b[2048], sCb[128];
    for (int idx = threadIdx.x; idx < 1024; idx += 256) { sWih[idx] = Wih[idx]; sWhh[idx] = Whh[idx]; }
    for (int idx = threadIdx.x; idx < 2048; idx += 256) {
        int m = idx >> 4, k = idx & 15;
        sW1b[idx] = W1[m * 32 + 16 + k];
    }
    for (int idx = threadIdx.x; idx < 288; idx += 256) sDW[idx] = decW[idx];
    if (threadIdx.x < 128) sCb[threadIdx.x] = cb[threadIdx.x];
    if (threadIdx.x < 64) { sbi[threadIdx.x] = bih[threadIdx.x]; sbh[threadIdx.x] = bhh[threadIdx.x]; }
    if (threadIdx.x < 16) sDB[threadIdx.x] = decB[threadIdx.x];
    __syncthreads();
    int i = blockIdx.x * 256 + threadIdx.x;
    float cx[16], h[16], c[16];
    #pragma unroll
    for (int k = 0; k < 16; ++k) {
        cx[k] = ws[WS_CTX + i * 16 + k];
        h[k]  = ws[WS_H   + i * 16 + k];
        c[k]  = ws[WS_C   + i * 16 + k];
    }
    float o0 = ws[WS_OUT + i * 2 + 0], o1 = ws[WS_OUT + i * 2 + 1];
    float e[16];
    for (int k = 0; k < 16; ++k) {
        float s = sDB[k] + sDW[k * 18 + 16] * o0 + sDW[k * 18 + 17] * o1;
        #pragma unroll
        for (int m = 0; m < 16; ++m) s += sDW[k * 18 + m] * cx[m];
        e[k] = fmaxf(s, 0.f);
    }
    float nh[16], nc[16];
    for (int u = 0; u < 16; ++u) {
        float gi = sbi[u]      + sbh[u];
        float gf = sbi[16 + u] + sbh[16 + u];
        float gg = sbi[32 + u] + sbh[32 + u];
        float go = sbi[48 + u] + sbh[48 + u];
        #pragma unroll
        for (int k = 0; k < 16; ++k) {
            gi += sWih[u * 16 + k]        * e[k] + sWhh[u * 16 + k]        * h[k];
            gf += sWih[(16 + u) * 16 + k] * e[k] + sWhh[(16 + u) * 16 + k] * h[k];
            gg += sWih[(32 + u) * 16 + k] * e[k] + sWhh[(32 + u) * 16 + k] * h[k];
            go += sWih[(48 + u) * 16 + k] * e[k] + sWhh[(48 + u) * 16 + k] * h[k];
        }
        nc[u] = sigm(gf) * c[u] + sigm(gi) * tanhf(gg);
        nh[u] = sigm(go) * tanhf(nc[u]);
    }
    #pragma unroll
    for (int k = 0; k < 16; ++k) {
        ws[WS_H + i * 16 + k] = nh[k];
        ws[WS_C + i * 16 + k] = nc[k];
    }
    // Hj[i] = pred_h[i] @ W1b^T + cb   (for pn pooling)
    for (int m = 0; m < 128; ++m) {
        float s = sCb[m];
        #pragma unroll
        for (int k = 0; k < 16; ++k) s += nh[k] * sW1b[m * 16 + k];
        ws[WS_HJ + i * 128 + m] = s;
    }
}

// ---------------- fused masked-max pooling with WMMA ---------------------
// For agent i (one block): pooled[i] = max_{j: nei>0} relu( relu(corr_ij@Wc + Hj[j]) @ W2^T + b2 )
__global__ __launch_bounds__(256) void tg_pool(
    const float* __restrict__ pos, const int* __restrict__ nei,
    const float* __restrict__ Hj,  const float* __restrict__ Wc,
    const float* __restrict__ W2,  const float* __restrict__ b2,
    float* __restrict__ outp)
{
    __shared__ float sWc[256];
    __shared__ int   sNei[512];
    __shared__ float sPart[256];
    __shared__ float sPos[2];

    const int i    = blockIdx.x;
    const int tid  = threadIdx.x;
    const int wave = tid >> 5;
    const int lane = tid & 31;
    const int col  = lane & 15;   // output column / A row index
    const int hi   = lane >> 4;   // lane-group

    sWc[tid] = Wc[tid];
    for (int j = tid; j < 512; j += 256) sNei[j] = nei[i * 512 + j];
    if (tid == 0) { sPos[0] = pos[2 * i]; sPos[1] = pos[2 * i + 1]; }
    __syncthreads();

    // B = W2^T in WMMA B-layout, 4 chunks of K=32, resident in VGPRs
    v16bf B[4];
    #pragma unroll
    for (int c = 0; c < 4; ++c) {
        #pragma unroll
        for (int v = 0; v < 8; ++v) {
            #pragma unroll
            for (int hh = 0; hh < 2; ++hh) {
                int k = 32 * c + hi * 16 + 2 * v + hh;
                B[c][2 * v + hh] = (__bf16)W2[col * 128 + k];
            }
        }
    }
    const float bn = b2[col];
    const float px = sPos[0], py = sPos[1];

    float part = 0.0f;  // relu output >= 0 and no-neighbor => 0: matches reference masking

    for (int tile = 0; tile < 4; ++tile) {
        const int j0 = (wave * 4 + tile) * 16;
        const int jr = j0 + col;  // j-row this lane builds A elements for
        const float dx = px - pos[2 * jr];
        const float dy = py - pos[2 * jr + 1];
        const float* hjrow = Hj + jr * 128;

        v8f acc = {};
        #pragma unroll
        for (int c = 0; c < 4; ++c) {
            v16bf A;
            #pragma unroll
            for (int p = 0; p < 8; ++p) {
                const int kb = 2 * p;
                const int k0 = 32 * c + kb + 16 * hi;
                const float2 hj = *(const float2*)(hjrow + k0);
                float v0 = fmaxf(fmaf(dx, sWc[k0],     fmaf(dy, sWc[128 + k0],     hj.x)), 0.f);
                float v1 = fmaxf(fmaf(dx, sWc[k0 + 1], fmaf(dy, sWc[128 + k0 + 1], hj.y)), 0.f);
                A[2 * p]     = (__bf16)v0;
                A[2 * p + 1] = (__bf16)v1;
            }
            acc = __builtin_amdgcn_wmma_f32_16x16x32_bf16(
                false, A, false, B[c], (short)0, acc, false, false);
        }
        // D layout: VGPR r -> row (r + 8*hi), lane -> column
        #pragma unroll
        for (int r = 0; r < 8; ++r) {
            const int jl = j0 + r + 8 * hi;
            const float w = fmaxf(acc[r] + bn, 0.f);
            if (sNei[jl] > 0) part = fmaxf(part, w);
        }
    }
    sPart[tid] = part;
    __syncthreads();
    if (tid < 16) {
        float m = 0.0f;
        #pragma unroll
        for (int w = 0; w < 8; ++w) {
            m = fmaxf(m, sPart[w * 32 + tid]);
            m = fmaxf(m, sPart[w * 32 + 16 + tid]);
        }
        outp[i * 16 + tid] = m;
    }
}

// ---------------- mid: mu/logscale, pos_s, Hj(pnc) ----------------------
__global__ __launch_bounds__(256) void tg_mid(
    const float* __restrict__ h2pW, const float* __restrict__ h2pB,
    const float* __restrict__ W1, const float* __restrict__ cb,
    float* __restrict__ ws, float* __restrict__ out, int t)
{
    __shared__ float sW1b[2048], sCb[128], sHW[64], sHB[4];
    for (int idx = threadIdx.x; idx < 2048; idx += 256) {
        int m = idx >> 4, k = idx & 15;
        sW1b[idx] = W1[m * 32 + 16 + k];
    }
    if (threadIdx.x < 128) sCb[threadIdx.x] = cb[threadIdx.x];
    if (threadIdx.x < 64)  sHW[threadIdx.x] = h2pW[threadIdx.x];
    if (threadIdx.x < 4)   sHB[threadIdx.x] = h2pB[threadIdx.x];
    __syncthreads();
    int i = blockIdx.x * 256 + threadIdx.x;
    float h[16], cc[16];
    #pragma unroll
    for (int k = 0; k < 16; ++k) {
        h[k]  = ws[WS_H + i * 16 + k];
        cc[k] = ws[WS_CTX + i * 16 + k] + h[k] + ws[WS_RG + i * 16 + k];
    }
    float ms[4];
    #pragma unroll
    for (int o = 0; o < 4; ++o) {
        float s = sHB[o];
        #pragma unroll
        for (int k = 0; k < 16; ++k) s += sHW[o * 16 + k] * cc[k];
        ms[o] = s;
    }
    float mu0 = ms[0], mu1 = ms[1];
    float s0 = fminf(fmaxf(ms[2], -9.f), 4.f);
    float s1 = fminf(fmaxf(ms[3], -9.f), 4.f);
    out[12288 + t * 1024 + i * 2 + 0] = mu0;   // mu section
    out[12288 + t * 1024 + i * 2 + 1] = mu1;
    out[24576 + t * 1024 + i * 2 + 0] = s0;    // logscale section
    out[24576 + t * 1024 + i * 2 + 1] = s1;
    ws[WS_SAMP + i * 2 + 0] = mu0;
    ws[WS_SAMP + i * 2 + 1] = mu1;
    ws[WS_POSS + i * 2 + 0] = ws[WS_POS + i * 2 + 0] + mu0;
    ws[WS_POSS + i * 2 + 1] = ws[WS_POS + i * 2 + 1] + mu1;
    // Hj for pnc pooling (overwrites WS_HJ; pn pooling already done)
    for (int m = 0; m < 128; ++m) {
        float s = sCb[m];
        #pragma unroll
        for (int k = 0; k < 16; ++k) s += h[k] * sW1b[m * 16 + k];
        ws[WS_HJ + i * 128 + m] = s;
    }
}

// ---------------- final: out_pred, pos update, carries -------------------
__global__ __launch_bounds__(256) void tg_final(
    const float* __restrict__ corrW, const float* __restrict__ corrB,
    float* __restrict__ ws, float* __restrict__ out, int t)
{
    int i = blockIdx.x * 256 + threadIdx.x;
    float s0 = corrB[0], s1 = corrB[1];
    #pragma unroll
    for (int k = 0; k < 16; ++k) {
        float cv = ws[WS_CTXC + i * 16 + k];
        s0 += corrW[k] * cv;
        s1 += corrW[16 + k] * cv;
    }
    s0 += ws[WS_SAMP + i * 2 + 0];
    s1 += ws[WS_SAMP + i * 2 + 1];
    out[t * 1024 + i * 2 + 0] = s0;            // pred section
    out[t * 1024 + i * 2 + 1] = s1;
    ws[WS_POS + i * 2 + 0] += s0;
    ws[WS_POS + i * 2 + 1] += s1;
    ws[WS_OUT + i * 2 + 0] = s0;
    ws[WS_OUT + i * 2 + 1] = s1;
}

extern "C" void kernel_launch(void* const* d_in, const int* in_sizes, int n_in,
                              void* d_out, int out_size, void* d_ws, size_t ws_size,
                              hipStream_t stream)
{
    (void)in_sizes; (void)n_in; (void)out_size; (void)ws_size;
    const float* traj_rel = (const float*)d_in[0];
    const float* obs_pos  = (const float*)d_in[1];
    const float* goal     = (const float*)d_in[3];
    const float* h0       = (const float*)d_in[4];
    const float* c0       = (const float*)d_in[5];
    const int*   nei      = (const int*)  d_in[6];
    const float* encW = (const float*)d_in[9],  *encB = (const float*)d_in[10];
    const float* goalW= (const float*)d_in[11], *goalB= (const float*)d_in[12];
    const float* decW = (const float*)d_in[13], *decB = (const float*)d_in[14];
    const float* h2pW = (const float*)d_in[15], *h2pB = (const float*)d_in[16];
    const float* corrW= (const float*)d_in[17], *corrB= (const float*)d_in[18];
    const float* tlWih= (const float*)d_in[19], *tlbih= (const float*)d_in[20];
    const float* tlWhh= (const float*)d_in[21], *tlbhh= (const float*)d_in[22];
    const float* plWih= (const float*)d_in[23], *plbih= (const float*)d_in[24];
    const float* plWhh= (const float*)d_in[25], *plbhh= (const float*)d_in[26];
    const float* pn_spW = (const float*)d_in[27], *pn_spb = (const float*)d_in[28];
    const float* pn_W1  = (const float*)d_in[29], *pn_b1  = (const float*)d_in[30];
    const float* pn_W2  = (const float*)d_in[31], *pn_b2  = (const float*)d_in[32];
    const float* pnc_spW= (const float*)d_in[33], *pnc_spb= (const float*)d_in[34];
    const float* pnc_W1 = (const float*)d_in[35], *pnc_b1 = (const float*)d_in[36];
    const float* pnc_W2 = (const float*)d_in[37], *pnc_b2 = (const float*)d_in[38];
    float* ws  = (float*)d_ws;
    float* out = (float*)d_out;

    tg_init<<<2, 256, 0, stream>>>(traj_rel, obs_pos, goal, goalW, goalB, ws);
    tg_fold<<<1, 256, 0, stream>>>(pn_spW, pn_spb, pn_W1, pn_b1,
                                   pnc_spW, pnc_spb, pnc_W1, pnc_b1, ws);
    tg_enc<<<2, 256, 0, stream>>>(traj_rel, h0, c0, encW, encB,
                                  tlWih, tlbih, tlWhh, tlbhh, ws);

    for (int t = 0; t < 12; ++t) {
        const int* nei_t = nei + (size_t)t * N_AG * N_AG;
        tg_dec<<<2, 256, 0, stream>>>(decW, decB, plWih, plbih, plWhh, plbhh,
                                      pn_W1, ws + WS_CBPN, ws);
        tg_pool<<<N_AG, 256, 0, stream>>>(ws + WS_POS, nei_t, ws + WS_HJ,
                                          ws + WS_WCPN, pn_W2, pn_b2, ws + WS_CTX);
        tg_mid<<<2, 256, 0, stream>>>(h2pW, h2pB, pnc_W1, ws + WS_CBPNC, ws, out, t);
        tg_pool<<<N_AG, 256, 0, stream>>>(ws + WS_POSS, nei_t, ws + WS_HJ,
                                          ws + WS_WCPNC, pnc_W2, pnc_b2, ws + WS_CTXC);
        tg_final<<<2, 256, 0, stream>>>(corrW, corrB, ws, out, t);
    }
}
